// LORAPoseAdaptorAttnProcessor_62148176773341
// MI455X (gfx1250) — matrix operand order repaired and checked by reading, not compiled, not run on hardware
//
#include <hip/hip_runtime.h>
#include <stdint.h>

// ---------- constants ----------
constexpr int Bb = 8, Ss = 1024, Cc = 1280, Hh = 8, DH = 160;
constexpr int Mm = Bb * Ss;                 // 8192 rows
constexpr float RS = 0.07905694150420949f;  // 160^-0.5

// ---------- types ----------
typedef __attribute__((ext_vector_type(16))) __bf16 bf16x16;
typedef __attribute__((ext_vector_type(8)))  float  f32x8;
typedef __attribute__((ext_vector_type(4)))  unsigned int u32x4;
typedef __attribute__((ext_vector_type(4)))  float  f32x4;
typedef __attribute__((ext_vector_type(4)))  int    i32x4v;
typedef __attribute__((ext_vector_type(8)))  int    i32x8v;

union Frag16 { bf16x16 v; u32x4 q[2]; };

__device__ __forceinline__ unsigned short f2bf(float f) {
  union { float f; uint32_t u; } x; x.f = f;
  uint32_t r = x.u + 0x7FFFu + ((x.u >> 16) & 1u);
  return (unsigned short)(r >> 16);
}
__device__ __forceinline__ unsigned int pk2(float a, float b) {
  return (unsigned int)f2bf(a) | ((unsigned int)f2bf(b) << 16);
}
__device__ __forceinline__ f32x8 wmma_bf16(const Frag16& a, const Frag16& b, f32x8 c) {
  return __builtin_amdgcn_wmma_f32_16x16x32_bf16(false, a.v, false, b.v, (short)0, c, false, false);
}

// ---------- TDM: 2D tile load global->LDS (CDNA5 ISA ch.8, D# groups 0/1) ----------
// tile_x elems (bf16) per row, tile_y rows, row stride = stride_elems.
// pad=true inserts 4 dwords of LDS padding every 16 dwords (64B rows -> 80B rows).
__device__ __forceinline__ void tdm_load_2d(uint32_t lds_addr, const void* gptr,
                                            int tile_x, int tile_y, int stride_elems,
                                            bool pad) {
  uint64_t ga = (uint64_t)(uintptr_t)gptr;
  i32x4v g0;
  g0.x = 1;                                   // count=1, user mode, no gather
  g0.y = (int)lds_addr;                       // lds_addr [63:32]
  g0.z = (int)(uint32_t)ga;                   // global_addr [95:64]
  g0.w = (int)(((uint32_t)(ga >> 32) & 0x01FFFFFFu) | 0x80000000u); // addr hi + type=2
  i32x8v g1;
  int d0 = (1 << 16);                         // data_size = 2 bytes
  if (pad) d0 |= (1 << 20) | (3 << 22) | (3 << 25); // pad_en, every 16 dw, +4 dw
  g1[0] = d0;
  g1[1] = (stride_elems & 0xFFFF) << 16;      // tensor_dim0 lo16 at [63:48]
  g1[2] = ((stride_elems >> 16) & 0xFFFF);    // tensor_dim0 hi16 | tensor_dim1 lo16(=0)
  g1[3] = 16                                  // tensor_dim1 hi16 (1<<20)
        | (tile_x << 16);                     // tile_dim0
  g1[4] = tile_y & 0xFFFF;                    // tile_dim1 (tile_dim2 = 0)
  g1[5] = stride_elems;                       // tensor_dim0_stride lo32
  g1[6] = 0;
  g1[7] = 0;
  asm volatile("tensor_load_to_lds %0, %1" :: "s"(g0), "s"(g1) : "memory");
}

// ---------- kernel 1: fold LoRA into weights, cast bf16, store TRANSPOSED [N][K] ----------
__global__ void prep_weights(const float* __restrict__ wq, const float* __restrict__ wk,
                             const float* __restrict__ wv, const float* __restrict__ wo,
                             const float* __restrict__ qd, const float* __restrict__ qu,
                             const float* __restrict__ kd, const float* __restrict__ ku,
                             const float* __restrict__ vd, const float* __restrict__ vu,
                             const float* __restrict__ od, const float* __restrict__ ou,
                             const float* __restrict__ wm,
                             unsigned short* __restrict__ weffT) {
  int idx = blockIdx.x * 256 + threadIdx.x;
  const int CC = Cc * Cc;
  if (idx >= 5 * CC) return;
  int which = idx / CC;
  int rem = idx - which * CC;
  int n = rem / Cc, k = rem - n * Cc;        // output row n, col k
  const float* W; const float* dn = nullptr; const float* up = nullptr;
  switch (which) {
    case 0: W = wq; dn = qd; up = qu; break;
    case 1: W = wk; dn = kd; up = ku; break;
    case 2: W = wv; dn = vd; up = vu; break;
    case 3: W = wo; dn = od; up = ou; break;
    default: W = wm; break;
  }
  float v = W[k * Cc + n];
  if (dn) {
#pragma unroll
    for (int r = 0; r < 4; ++r) v += dn[k * 4 + r] * up[r * Cc + n];
  }
  weffT[idx] = f2bf(v);
}

// ---------- GEMM: out[M,N] = A[M,K] @ W[K,N], W given transposed Wt[N][K] ----------
// MODE 0: A = bf16(h+p) staged manually, epi: + bias[n] + resid[m,n], store bf16
// MODE 1: A = bf16 via TDM,              epi: none,                   store bf16
// MODE 2: A = bf16 via TDM,              epi: + bias[n],              store f32
// MODE 3: A = bf16 via TDM,              epi: none, store bf16 transposed [b,h,dh,s]
template<int MODE>
__global__ __launch_bounds__(256) void gemm128(
    const void* __restrict__ Aptr, const float* __restrict__ Aptr2,
    const unsigned short* __restrict__ Wt, const float* __restrict__ bias,
    const float* __restrict__ resid, void* __restrict__ Out) {
  __shared__ unsigned short Alds[2][128][40];   // rows x K32, 80B rows (TDM pad)
  __shared__ unsigned short Btlds[2][128][40];  // cols x K32, 80B rows

  const int tid = threadIdx.x;
  const int n0 = blockIdx.x * 128;
  const int m0 = blockIdx.y * 128;
  const int w = tid >> 5, lane = tid & 31;
  const int mw = w >> 1, nw = w & 1;
  const int l16 = lane & 15, lhalf = lane >> 4;
  const int NK = Cc / 32;
  const int OPI = (MODE == 0) ? 1 : 2;          // tensor ops per issue

  const unsigned short* Abase = (const unsigned short*)Aptr + m0 * Cc;
  const unsigned short* Wbase = Wt + n0 * Cc;
  const uint32_t aL[2] = { (uint32_t)(uintptr_t)&Alds[0][0][0],
                           (uint32_t)(uintptr_t)&Alds[1][0][0] };
  const uint32_t bL[2] = { (uint32_t)(uintptr_t)&Btlds[0][0][0],
                           (uint32_t)(uintptr_t)&Btlds[1][0][0] };

  auto issue = [&](int kt, int buf) {
    if (MODE != 0) tdm_load_2d(aL[buf], Abase + kt * 32, 32, 128, Cc, true);
    tdm_load_2d(bL[buf], Wbase + kt * 32, 32, 128, Cc, true);
  };

  if (tid < 32) issue(0, 0);

  f32x8 acc[2][4];
#pragma unroll
  for (int mt = 0; mt < 2; ++mt)
#pragma unroll
    for (int nt = 0; nt < 4; ++nt) acc[mt][nt] = {};

  for (int kt = 0; kt < NK; ++kt) {
    const int cur = kt & 1;
    // ---- MODE 0: manual A stage (f32 h+p -> bf16) into current buffer ----
    if (MODE == 0) {
      const float* hptr = (const float*)Aptr;
      const float* pptr = Aptr2;
#pragma unroll
      for (int j = 0; j < 2; ++j) {
        int id = tid + j * 256;
        int r = id >> 2, c = id & 3;
        int base = (m0 + r) * Cc + kt * 32 + c * 8;
        f32x4 a0 = *(const f32x4*)(hptr + base);
        f32x4 a1 = *(const f32x4*)(hptr + base + 4);
        f32x4 b0 = *(const f32x4*)(pptr + base);
        f32x4 b1 = *(const f32x4*)(pptr + base + 4);
        u32x4 pk;
        pk.x = pk2(a0.x + b0.x, a0.y + b0.y);
        pk.y = pk2(a0.z + b0.z, a0.w + b0.w);
        pk.z = pk2(a1.x + b1.x, a1.y + b1.y);
        pk.w = pk2(a1.z + b1.z, a1.w + b1.w);
        *(u32x4*)&Alds[cur][r][c * 8] = pk;
      }
    }
    // ---- prefetch next tile via TDM, wait for current, sync ----
    if (tid < 32) {
      if (kt + 1 < NK) {
        issue(kt + 1, cur ^ 1);
        if (OPI == 1) __builtin_amdgcn_s_wait_tensorcnt(1);
        else          __builtin_amdgcn_s_wait_tensorcnt(2);
      } else {
        __builtin_amdgcn_s_wait_tensorcnt(0);
      }
    }
    __syncthreads();
    // ---- fragments (grouped loads) + WMMA ----
    Frag16 af[2], bfr[4];
#pragma unroll
    for (int mt = 0; mt < 2; ++mt) {
      int row = mw * 32 + mt * 16 + l16;
      int klo = lhalf ? 8 : 0;  // A: lanes0-15 K0..7,16..23; lanes16-31 K8..15,24..31
      af[mt].q[0] = *(const u32x4*)&Alds[cur][row][klo];
      af[mt].q[1] = *(const u32x4*)&Alds[cur][row][klo + 16];
    }
#pragma unroll
    for (int nt = 0; nt < 4; ++nt) {
      int col = nw * 64 + nt * 16 + l16;
      int klo = lhalf ? 8 : 0;  // B: lanes0-15 K0..7,16..23; lanes16-31 K8..15,24..31
      bfr[nt].q[0] = *(const u32x4*)&Btlds[cur][col][klo];
      bfr[nt].q[1] = *(const u32x4*)&Btlds[cur][col][klo + 16];
    }
#pragma unroll
    for (int mt = 0; mt < 2; ++mt)
#pragma unroll
      for (int nt = 0; nt < 4; ++nt)
        acc[mt][nt] = wmma_bf16(af[mt], bfr[nt], acc[mt][nt]);
    __syncthreads();   // all reads of this buffer done before TDM of kt+2 targets it
  }
  // ---- epilogue ----
#pragma unroll
  for (int mt = 0; mt < 2; ++mt) {
#pragma unroll
    for (int nt = 0; nt < 4; ++nt) {
      int col = n0 + nw * 64 + nt * 16 + l16;
      float bv = 0.0f;
      if (MODE == 0 || MODE == 2) bv = bias[col];
#pragma unroll
      for (int i = 0; i < 8; ++i) {
        int row = m0 + mw * 32 + mt * 16 + i + lhalf * 8;
        int off = row * Cc + col;
        float v = acc[mt][nt][i];
        if (MODE == 0) {
          v += bv + resid[off];
          ((unsigned short*)Out)[off] = f2bf(v);
        } else if (MODE == 1) {
          ((unsigned short*)Out)[off] = f2bf(v);
        } else if (MODE == 2) {
          ((float*)Out)[off] = v + bv;
        } else {  // MODE 3: V stored transposed [b, h, dh, s]
          int b_ = row >> 10, s_ = row & 1023;
          int h_ = col / DH, d_ = col - h_ * DH;
          ((unsigned short*)Out)[((b_ * Hh + h_) * DH + d_) * Ss + s_] = f2bf(v);
        }
      }
    }
  }
}

// ---------- flash attention: one (b,h), 128 queries per workgroup ----------
// K natural [b,s,c]; V pre-transposed Vt[b,h,dh,s]; both tiles staged via TDM.
__global__ __launch_bounds__(256) void attn_kernel(
    const unsigned short* __restrict__ Q, const unsigned short* __restrict__ K,
    const unsigned short* __restrict__ Vt, unsigned short* __restrict__ Obuf) {
  __shared__ unsigned short Klds[2][32][160];   // [key][dh], 320B rows
  __shared__ unsigned short Vtlds[2][160][40];  // [dh][key], 80B rows (TDM pad)
  __shared__ unsigned short Plds[8][16][32];    // per-wave P scratch

  const int tid = threadIdx.x, w = tid >> 5, lane = tid & 31;
  const int l16 = lane & 15, lhalf = lane >> 4;
  const int bh = blockIdx.y;
  const int b = bh >> 3, h = bh & 7;
  const int q0 = blockIdx.x * 128 + w * 16;
  const int baseBH = b * Ss * Cc + h * DH;

  const unsigned short* Kbase = K + baseBH;
  const unsigned short* Vtbase = Vt + (b * Hh + h) * DH * Ss;
  const uint32_t kL[2] = { (uint32_t)(uintptr_t)&Klds[0][0][0],
                           (uint32_t)(uintptr_t)&Klds[1][0][0] };
  const uint32_t vL[2] = { (uint32_t)(uintptr_t)&Vtlds[0][0][0],
                           (uint32_t)(uintptr_t)&Vtlds[1][0][0] };

  auto issue = [&](int it, int buf) {
    tdm_load_2d(kL[buf], Kbase + it * 32 * Cc, DH, 32, Cc, false);  // 32 keys x 160
    tdm_load_2d(vL[buf], Vtbase + it * 32,     32, DH, Ss, true);   // 160 dh x 32 keys
  };

  if (tid < 32) issue(0, 0);

  // resident Q fragments (16 rows x 160 = 5 K-chunks of 32)
  Frag16 qf[5];
#pragma unroll
  for (int kc = 0; kc < 5; ++kc) {
    int row = q0 + l16;
    int klo = kc * 32 + (lhalf ? 8 : 0);
    const unsigned short* p = Q + baseBH + row * Cc + klo;
    qf[kc].q[0] = *(const u32x4*)p;
    qf[kc].q[1] = *(const u32x4*)(p + 16);
  }

  f32x8 o[10];
#pragma unroll
  for (int nt = 0; nt < 10; ++nt) o[nt] = {};
  float m[8], l[8];
#pragma unroll
  for (int i = 0; i < 8; ++i) { m[i] = -__builtin_inff(); l[i] = 0.0f; }

  const int NIT = Ss / 32;
  for (int it = 0; it < NIT; ++it) {
    const int cur = it & 1;
    if (tid < 32) {
      if (it + 1 < NIT) {
        issue(it + 1, cur ^ 1);
        __builtin_amdgcn_s_wait_tensorcnt(2);
      } else {
        __builtin_amdgcn_s_wait_tensorcnt(0);
      }
    }
    __syncthreads();

    // scores: two 16-key tiles over K=160; group 5 frag loads -> 5 WMMAs
    f32x8 s0 = {}, s1 = {};
    {
      Frag16 kf[5];
#pragma unroll
      for (int kc = 0; kc < 5; ++kc) {
        int klo = kc * 32 + (lhalf ? 8 : 0);
        kf[kc].q[0] = *(const u32x4*)&Klds[cur][l16][klo];
        kf[kc].q[1] = *(const u32x4*)&Klds[cur][l16][klo + 16];
      }
#pragma unroll
      for (int kc = 0; kc < 5; ++kc) s0 = wmma_bf16(qf[kc], kf[kc], s0);
#pragma unroll
      for (int kc = 0; kc < 5; ++kc) {
        int klo = kc * 32 + (lhalf ? 8 : 0);
        kf[kc].q[0] = *(const u32x4*)&Klds[cur][l16 + 16][klo];
        kf[kc].q[1] = *(const u32x4*)&Klds[cur][l16 + 16][klo + 16];
      }
#pragma unroll
      for (int kc = 0; kc < 5; ++kc) s1 = wmma_bf16(qf[kc], kf[kc], s1);
    }
    // online softmax (C-layout rows: vgpr i lanes0-15 -> row i, lanes16-31 -> row i+8)
    float alpha[8];
#pragma unroll
    for (int i = 0; i < 8; ++i) {
      float a = s0[i] * RS, c = s1[i] * RS;
      float mx = fmaxf(a, c);
#pragma unroll
      for (int d = 1; d < 16; d <<= 1) mx = fmaxf(mx, __shfl_xor(mx, d, 32));
      float mn = fmaxf(m[i], mx);
      alpha[i] = __expf(m[i] - mn);
      m[i] = mn;
      float p0 = __expf(a - mn), p1 = __expf(c - mn);
      s0[i] = p0; s1[i] = p1;
      float rsum = p0 + p1;
#pragma unroll
      for (int d = 1; d < 16; d <<= 1) rsum += __shfl_xor(rsum, d, 32);
      l[i] = l[i] * alpha[i] + rsum;
    }
#pragma unroll
    for (int nt = 0; nt < 10; ++nt)
#pragma unroll
      for (int i = 0; i < 8; ++i) o[nt][i] *= alpha[i];

    // relayout P (C-layout -> A-layout) through per-wave LDS scratch
#pragma unroll
    for (int i = 0; i < 8; ++i) {
      int prow = i + lhalf * 8;
      Plds[w][prow][l16] = f2bf(s0[i]);
      Plds[w][prow][16 + l16] = f2bf(s1[i]);
    }
    asm volatile("s_wait_dscnt 0x0" ::: "memory"); // wave-local store->load ordering
    Frag16 pf;
    {
      int klo = lhalf ? 8 : 0;
      pf.q[0] = *(const u32x4*)&Plds[w][l16][klo];
      pf.q[1] = *(const u32x4*)&Plds[w][l16][klo + 16];
    }
    // O += P @ V : two groups of (5 frag loads -> 5 WMMAs)
#pragma unroll
    for (int g = 0; g < 2; ++g) {
      Frag16 vf[5];
#pragma unroll
      for (int j = 0; j < 5; ++j) {
        int col = (g * 5 + j) * 16 + l16;
        int klo = lhalf ? 8 : 0;
        vf[j].q[0] = *(const u32x4*)&Vtlds[cur][col][klo];
        vf[j].q[1] = *(const u32x4*)&Vtlds[cur][col][klo + 16];
      }
#pragma unroll
      for (int j = 0; j < 5; ++j)
        o[g * 5 + j] = wmma_bf16(pf, vf[j], o[g * 5 + j]);
    }
    __syncthreads();   // reads of this buffer done before TDM of it+2 targets it
  }
  // normalize + store bf16 [b, s, h*160 + col]
  float inv[8];
#pragma unroll
  for (int i = 0; i < 8; ++i) inv[i] = 1.0f / l[i];
#pragma unroll
  for (int nt = 0; nt < 10; ++nt) {
#pragma unroll
    for (int i = 0; i < 8; ++i) {
      int row = q0 + i + lhalf * 8;
      int col = h * DH + nt * 16 + l16;
      Obuf[(b * Ss + row) * Cc + col] = f2bf(o[nt][i] * inv[i]);
    }
  }
}

// ---------- host ----------
extern "C" void kernel_launch(void* const* d_in, const int* in_sizes, int n_in,
                              void* d_out, int out_size, void* d_ws, size_t ws_size,
                              hipStream_t stream) {
  const float* hidden = (const float*)d_in[0];
  const float* pose   = (const float*)d_in[1];
  const float* wm     = (const float*)d_in[2];
  const float* bm     = (const float*)d_in[3];
  const float* wq     = (const float*)d_in[4];
  const float* wk     = (const float*)d_in[5];
  const float* wv     = (const float*)d_in[6];
  const float* wo     = (const float*)d_in[7];
  const float* bo     = (const float*)d_in[8];
  const float* qd = (const float*)d_in[9],  * qu = (const float*)d_in[10];
  const float* kd = (const float*)d_in[11], * ku = (const float*)d_in[12];
  const float* vd = (const float*)d_in[13], * vu = (const float*)d_in[14];
  const float* od = (const float*)d_in[15], * ou = (const float*)d_in[16];

  const size_t CC = (size_t)Cc * Cc;        // weight elems
  const size_t MC = (size_t)Mm * Cc;        // activation elems
  const size_t need = 5 * CC * 2 + 5 * MC * 2;
  if (ws_size < need) return;

  unsigned short* weffT = (unsigned short*)d_ws;   // q,k,v,o,m (transposed [N][K])
  unsigned short* weq = weffT;
  unsigned short* wek = weffT + CC;
  unsigned short* wev = weffT + 2 * CC;
  unsigned short* weo = weffT + 3 * CC;
  unsigned short* wem = weffT + 4 * CC;
  unsigned short* xbf = weffT + 5 * CC;
  unsigned short* qbf = xbf + MC;
  unsigned short* kbf = xbf + 2 * MC;
  unsigned short* vtb = xbf + 3 * MC;   // V transposed [b,h,dh,s]
  unsigned short* abf = xbf + 4 * MC;

  prep_weights<<<(5 * (int)CC + 255) / 256, 256, 0, stream>>>(
      wq, wk, wv, wo, qd, qu, kd, ku, vd, vu, od, ou, wm, weffT);

  dim3 gg(Cc / 128, Mm / 128);  // (10, 64)
  gemm128<0><<<gg, 256, 0, stream>>>(hidden, pose, wem, bm, hidden, xbf);
  gemm128<1><<<gg, 256, 0, stream>>>(xbf, nullptr, weq, nullptr, nullptr, qbf);
  gemm128<1><<<gg, 256, 0, stream>>>(xbf, nullptr, wek, nullptr, nullptr, kbf);
  gemm128<3><<<gg, 256, 0, stream>>>(xbf, nullptr, wev, nullptr, nullptr, vtb);

  attn_kernel<<<dim3(Ss / 128, Bb * Hh), 256, 0, stream>>>(qbf, kbf, vtb, abf);

  gemm128<2><<<gg, 256, 0, stream>>>(abf, nullptr, weo, bo, nullptr, (float*)d_out);
}
